// TFAttention_41961830482672
// MI455X (gfx1250) — matrix operand order repaired and checked by reading, not compiled, not run on hardware
//
#include <hip/hip_runtime.h>
#include <hip/hip_bf16.h>

// ---------------------------------------------------------------------------
// Windowed attention on MI455X (gfx1250, wave32, WMMA + TDM).
//   B=8 S=4096 N=512 K=512 H=8 WS=128 HD=64  -> M = B*S = 32768 tokens
// Pipeline:
//   0) cvt        : x, Wq/Wk/Wv/Wp converted to f16 ONCE (halves GEMM read BW,
//                   removes all cvt work from GEMM inner loops)
//   1) gemm512 x3 : xh @ Wh^T + b -> q,k (f16 row-major), vT (f16 transposed);
//                   A/B tiles staged by the Tensor Data Mover, double-buffered
//   2) attn_win   : per window-head attention; tiles TDM-staged
//   3) gemm512    : attnOut(f16) @ Wp^T + bp -> out (f32)
// All matmuls use V_WMMA_F32_16X16X32_F16 (f32 accumulate).
// ---------------------------------------------------------------------------

typedef _Float16 half_t;
typedef __attribute__((ext_vector_type(16))) _Float16 v16h;
typedef __attribute__((ext_vector_type(8)))  _Float16 v8h;
typedef __attribute__((ext_vector_type(8)))  float    v8f;
typedef __attribute__((ext_vector_type(4)))  float    v4f;
typedef unsigned int u32x4 __attribute__((ext_vector_type(4)));
typedef int          i32x4 __attribute__((ext_vector_type(4)));
typedef int          i32x8 __attribute__((ext_vector_type(8)));

#define TOKENS   32768   // B*S
#define DIM      512     // N == K_DIM
#define WINSZ    128
#define HEADDIM  64
#define NHEADS   8

// A-fragment (16x32 f16): lane holds row M=lane&15; lanes<16 take K[0..7]+K[16..23]
// (v8h chunks 0 and 2), lanes>=16 take chunks 1 and 3.  (ISA 7.12.2)
__device__ __forceinline__ v16h make_a_frag(const half_t* row32, int laneHi) {
  const v8h* p = (const v8h*)row32;
  v8h lo = p[laneHi];
  v8h hi = p[laneHi + 2];
  v16h r;
#pragma unroll
  for (int i = 0; i < 8; ++i) { r[i] = lo[i]; r[i + 8] = hi[i]; }
  return r;
}

// B-fragment (32x16 f16): lane holds column N=lane&15; same K order as A:
// lanes<16 take K[0..7]+K[16..23] (v8h chunks 0 and 2), lanes>=16 chunks 1 and 3.
__device__ __forceinline__ v16h make_b_frag(const half_t* btrow32, int laneHi) {
  const v8h* p = (const v8h*)btrow32;
  v8h lo = p[laneHi];
  v8h hi = p[laneHi + 2];
  v16h r;
#pragma unroll
  for (int i = 0; i < 8; ++i) { r[i] = lo[i]; r[i + 8] = hi[i]; }
  return r;
}

__device__ __forceinline__ v8f wmma_f16(v16h a, v16h b, v8f c) {
  return __builtin_amdgcn_wmma_f32_16x16x32_f16(false, a, false, b, (short)0, c,
                                                false, false);
}

#if __has_builtin(__builtin_amdgcn_tensor_load_to_lds)
#define USE_TDM 1
// Issue one TDM 2D tile load (wave-level op; EXEC ignored).  Units:
// global_addr/lds_addr in bytes, dims/strides in data_size (=2B) elements.
// D# packing per CDNA5 ISA 8.3/8.4.
__device__ __forceinline__ void tdm_load_2d(unsigned lds_off, const void* gaddr,
                                            unsigned tile_w, unsigned tile_h,
                                            unsigned stride_elems,
                                            unsigned tensor_w, unsigned tensor_h) {
  unsigned long long ga = (unsigned long long)gaddr;
  u32x4 g0;
  g0[0] = 1u;                                   // count=1 (valid user D#)
  g0[1] = lds_off;                              // lds_addr (bytes)
  g0[2] = (unsigned)ga;                         // global_addr[31:0]
  g0[3] = (unsigned)(ga >> 32) | (2u << 30);    // global_addr[56:32] | type=2
  i32x8 g1;
  g1[0] = (int)(1u << 16);                      // data_size=1 -> 2 bytes
  g1[1] = (int)((tensor_w & 0xFFFFu) << 16);    // tensor_dim0[15:0]
  g1[2] = (int)(((tensor_w >> 16) & 0xFFFFu) | ((tensor_h & 0xFFFFu) << 16));
  g1[3] = (int)(((tensor_h >> 16) & 0xFFFFu) | (tile_w << 16));  // tile_dim0
  g1[4] = (int)(tile_h & 0xFFFFu);              // tile_dim1 (tile_dim2=0)
  g1[5] = (int)stride_elems;                    // tensor_dim0_stride[31:0]
  g1[6] = 0;
  g1[7] = 0;
  i32x4 z4 = {0, 0, 0, 0};
#if __clang_major__ >= 23
  i32x8 z8 = {0, 0, 0, 0, 0, 0, 0, 0};
  __builtin_amdgcn_tensor_load_to_lds(g0, g1, z4, z4, z8, 0);
#else
  __builtin_amdgcn_tensor_load_to_lds(g0, g1, z4, z4, 0);
#endif
}
__device__ __forceinline__ unsigned lds_offset_of(const void* p) {
  return (unsigned)(unsigned long long)p;       // generic LDS addr: low 32 = offset
}
#else
#define USE_TDM 0
#warning "tensor_load_to_lds builtin unavailable - using per-lane staging fallback"
#endif

// ---------------------------------------------------------------------------
// One-time f32 -> f16 conversion (x and the four weight matrices).
// ---------------------------------------------------------------------------
__global__ __launch_bounds__(256) void cvt_f32_f16(const float* __restrict__ src,
                                                   half_t* __restrict__ dst, int n) {
  const int i = (blockIdx.x * 256 + threadIdx.x) * 8;
  if (i < n) {
    v4f a = *(const v4f*)(src + i);
    v4f b = *(const v4f*)(src + i + 4);
    v8h o;
    o[0] = (half_t)a[0]; o[1] = (half_t)a[1]; o[2] = (half_t)a[2]; o[3] = (half_t)a[3];
    o[4] = (half_t)b[0]; o[5] = (half_t)b[1]; o[6] = (half_t)b[2]; o[7] = (half_t)b[3];
    *(v8h*)(dst + i) = o;
  }
}

// ---------------------------------------------------------------------------
// M x 512 x 512 GEMM: C = A @ W^T + bias, all-f16 operands, f32 accumulate.
//   mode 0: out f16 row-major (q,k)   mode 1: out f16 [win][h][d][r] (vT)
//   mode 2: out f32 row-major (final projection)
// Block: 256 threads (8 waves), tile 128(M) x 64(N) x 32(K).
// A/B tiles double-buffered in LDS, filled by TDM (TENSORcnt-tracked).
// ---------------------------------------------------------------------------
__global__ __launch_bounds__(256) void gemm512(
    const half_t* __restrict__ A, const half_t* __restrict__ Wh,
    const float* __restrict__ bias, half_t* __restrict__ outH,
    float* __restrict__ outF, int mode) {
  __shared__ __attribute__((aligned(16))) half_t As[2][128 * 32];  // 16 KB
  __shared__ __attribute__((aligned(16))) half_t Bs[2][64 * 32];   //  8 KB

  const int tid = threadIdx.x;
  const int lane = tid & 31, wave = tid >> 5;
  const int laneLo = lane & 15, laneHi = lane >> 4;
  const int m0 = blockIdx.x * 128;
  const int n0 = blockIdx.y * 64;

  v8f acc[4];
#pragma unroll
  for (int t = 0; t < 4; ++t) {
    v8f z = {0.f, 0.f, 0.f, 0.f, 0.f, 0.f, 0.f, 0.f};
    acc[t] = z;
  }

#if USE_TDM
  if (wave == 0) {  // prologue: stage K-chunk 0 into buffer 0
    tdm_load_2d(lds_offset_of(As[0]), A  + (size_t)m0 * DIM, 32, 128, DIM, DIM, TOKENS);
    tdm_load_2d(lds_offset_of(Bs[0]), Wh + (size_t)n0 * DIM, 32, 64,  DIM, DIM, DIM);
  }
#endif

  for (int kk = 0; kk < DIM; kk += 32) {
    const int cur = (kk >> 5) & 1;
#if USE_TDM
    if (wave == 0) {
      if (kk + 32 < DIM) {  // stage next K-chunk into the other buffer
        tdm_load_2d(lds_offset_of(As[cur ^ 1]), A  + (size_t)m0 * DIM + kk + 32,
                    32, 128, DIM, DIM, TOKENS);
        tdm_load_2d(lds_offset_of(Bs[cur ^ 1]), Wh + (size_t)n0 * DIM + kk + 32,
                    32, 64, DIM, DIM, DIM);
        __builtin_amdgcn_s_wait_tensorcnt(2);  // current tile done (in-order)
      } else {
        __builtin_amdgcn_s_wait_tensorcnt(0);
      }
    }
#else
    {  // fallback: per-lane staging
      const int ra = tid >> 1, ca = (tid & 1) * 16;
      const half_t* srca = A + (size_t)(m0 + ra) * DIM + kk + ca;
      *(v8h*)(As[cur] + ra * 32 + ca)     = *(const v8h*)(srca);
      *(v8h*)(As[cur] + ra * 32 + ca + 8) = *(const v8h*)(srca + 8);
      const int rb = tid >> 2, cb = (tid & 3) * 8;
      const half_t* srcb = Wh + (size_t)(n0 + rb) * DIM + kk + cb;
      *(v8h*)(Bs[cur] + rb * 32 + cb) = *(const v8h*)(srcb);
    }
#endif
    __syncthreads();

    v16h af = make_a_frag(As[cur] + (16 * wave + laneLo) * 32, laneHi);
    v16h bf[4];
#pragma unroll
    for (int t = 0; t < 4; ++t)
      bf[t] = make_b_frag(Bs[cur] + (16 * t + laneLo) * 32, laneHi);
#pragma unroll
    for (int t = 0; t < 4; ++t)
      acc[t] = wmma_f16(af, bf[t], acc[t]);
    __syncthreads();
  }

  // epilogue: bias + store.  C layout: row = 16*wave + j + 8*laneHi, col = 16t+laneLo
#pragma unroll
  for (int t = 0; t < 4; ++t) {
    const int n = n0 + 16 * t + laneLo;
    const float bv = bias[n];
#pragma unroll
    for (int j = 0; j < 8; ++j) {
      const int m = m0 + 16 * wave + j + 8 * laneHi;
      const float val = acc[t][j] + bv;
      if (mode == 0) {
        outH[(size_t)m * DIM + n] = (half_t)val;
      } else if (mode == 1) {
        const int win = m >> 7, r = m & 127;
        const int h = n >> 6, d = n & 63;
        outH[(((size_t)(win * NHEADS + h) * HEADDIM + d) << 7) + r] = (half_t)val;
      } else {
        outF[(size_t)m * DIM + n] = val;
      }
    }
  }
}

// ---------------------------------------------------------------------------
// Windowed attention: one block per (window, head).  8 waves; each wave owns
// 16 query rows.  Tiles staged by TDM (3 descriptors), tracked by TENSORcnt.
// LDS: q 16K + k 16K + vT 16K + probs 32K = 80 KB -> 4 workgroups per WGP.
// ---------------------------------------------------------------------------
__global__ __launch_bounds__(256) void attn_win(
    const half_t* __restrict__ q, const half_t* __restrict__ k,
    const half_t* __restrict__ vT, half_t* __restrict__ o) {
  __shared__ __attribute__((aligned(16))) half_t qs[128 * 64];
  __shared__ __attribute__((aligned(16))) half_t ks[128 * 64];
  __shared__ __attribute__((aligned(16))) half_t vs[64 * 128];
  __shared__ __attribute__((aligned(16))) half_t ps[128 * 128];

  const int tid = threadIdx.x;
  const int lane = tid & 31, wave = tid >> 5;
  const int laneLo = lane & 15, laneHi = lane >> 4;
  const int wh = blockIdx.x;
  const int win = wh >> 3, h = wh & 7;

#if USE_TDM
  if (wave == 0) {
    // q,k tiles: 64 x 128 (row stride 512 elems); vT tile: contiguous 8192 elems
    const half_t* qsrc = q + (size_t)(win * WINSZ) * DIM + h * HEADDIM;
    const half_t* ksrc = k + (size_t)(win * WINSZ) * DIM + h * HEADDIM;
    const half_t* vsrc = vT + (((size_t)(win * NHEADS + h) * HEADDIM) << 7);
    tdm_load_2d(lds_offset_of(qs), qsrc, HEADDIM, WINSZ, DIM, DIM, WINSZ);
    tdm_load_2d(lds_offset_of(ks), ksrc, HEADDIM, WINSZ, DIM, DIM, WINSZ);
    tdm_load_2d(lds_offset_of(vs), vsrc, 8192u, 1u, 8192u, 8192u, 1u);
    __builtin_amdgcn_s_wait_tensorcnt(0);
  }
#else
  {
    const int r = tid >> 1, c0 = (tid & 1) * 32;
    const half_t* qsrc = q + (size_t)(win * WINSZ + r) * DIM + h * HEADDIM + c0;
    const half_t* ksrc = k + (size_t)(win * WINSZ + r) * DIM + h * HEADDIM + c0;
#pragma unroll
    for (int i = 0; i < 32; i += 8) {
      *(v8h*)(qs + r * 64 + c0 + i) = *(const v8h*)(qsrc + i);
      *(v8h*)(ks + r * 64 + c0 + i) = *(const v8h*)(ksrc + i);
    }
    const int d = tid >> 2, r0 = (tid & 3) * 32;
    const half_t* vsrc = vT + (((size_t)(win * NHEADS + h) * HEADDIM + d) << 7) + r0;
#pragma unroll
    for (int i = 0; i < 32; i += 8)
      *(v8h*)(vs + d * 128 + r0 + i) = *(const v8h*)(vsrc + i);
  }
#endif
  __syncthreads();

  // S = q @ k^T  (gemm-K = HEADDIM = 64 -> 2 chunks of 32)
  v8f s[8];
#pragma unroll
  for (int t = 0; t < 8; ++t) {
    v8f z = {0.f, 0.f, 0.f, 0.f, 0.f, 0.f, 0.f, 0.f};
    s[t] = z;
  }
  const half_t* qrow = qs + (16 * wave + laneLo) * 64;
#pragma unroll
  for (int kc = 0; kc < 64; kc += 32) {
    v16h af = make_a_frag(qrow + kc, laneHi);
#pragma unroll
    for (int g = 0; g < 8; g += 4) {        // batch 4 B-frags, then 4 WMMAs
      v16h bf[4];
#pragma unroll
      for (int t = 0; t < 4; ++t)
        bf[t] = make_b_frag(ks + (16 * (g + t) + laneLo) * 64 + kc, laneHi);
#pragma unroll
      for (int t = 0; t < 4; ++t)
        s[g + t] = wmma_f16(af, bf[t], s[g + t]);
    }
  }

  // row softmax: for fixed j the row lives in the lane's 16-lane half.
  const float scale = 0.125f;  // HD^-0.5
#pragma unroll
  for (int j = 0; j < 8; ++j) {
    float mx = -1e30f;
#pragma unroll
    for (int t = 0; t < 8; ++t) mx = fmaxf(mx, s[t][j] * scale);
#pragma unroll
    for (int off = 1; off < 16; off <<= 1) mx = fmaxf(mx, __shfl_xor(mx, off, 32));
    float sum = 0.f;
#pragma unroll
    for (int t = 0; t < 8; ++t) {
      float e = __expf(s[t][j] * scale - mx);
      s[t][j] = e;
      sum += e;
    }
#pragma unroll
    for (int off = 1; off < 16; off <<= 1) sum += __shfl_xor(sum, off, 32);
    const float inv = 1.f / sum;
#pragma unroll
    for (int t = 0; t < 8; ++t) s[t][j] *= inv;
  }

  // spill probs to LDS as f16 A-operand for the PV gemm
#pragma unroll
  for (int t = 0; t < 8; ++t)
#pragma unroll
    for (int j = 0; j < 8; ++j)
      ps[(16 * wave + j + 8 * laneHi) * 128 + 16 * t + laneLo] = (half_t)s[t][j];
  __syncthreads();

  // O = P @ V  (gemm-K = WINSZ = 128 -> 4 chunks of 32)
  v8f ov[4];
#pragma unroll
  for (int t = 0; t < 4; ++t) {
    v8f z = {0.f, 0.f, 0.f, 0.f, 0.f, 0.f, 0.f, 0.f};
    ov[t] = z;
  }
  const half_t* prow = ps + (16 * wave + laneLo) * 128;
#pragma unroll
  for (int kc = 0; kc < 128; kc += 32) {
    v16h af = make_a_frag(prow + kc, laneHi);
    v16h bf[4];
#pragma unroll
    for (int t = 0; t < 4; ++t)
      bf[t] = make_b_frag(vs + (16 * t + laneLo) * 128 + kc, laneHi);
#pragma unroll
    for (int t = 0; t < 4; ++t)
      ov[t] = wmma_f16(af, bf[t], ov[t]);
  }

  // store attn output f16 row-major [token][512] (window reverse is implicit)
#pragma unroll
  for (int t = 0; t < 4; ++t) {
#pragma unroll
    for (int j = 0; j < 8; ++j) {
      const int m = win * WINSZ + 16 * wave + j + 8 * laneHi;
      const int col = h * HEADDIM + 16 * t + laneLo;
      o[(size_t)m * DIM + col] = (half_t)ov[t][j];
    }
  }
}

extern "C" void kernel_launch(void* const* d_in, const int* in_sizes, int n_in,
                              void* d_out, int out_size, void* d_ws, size_t ws_size,
                              hipStream_t stream) {
  const float* x  = (const float*)d_in[0];
  const float* Wq = (const float*)d_in[1];
  const float* bq = (const float*)d_in[2];
  const float* Wk = (const float*)d_in[3];
  const float* bk = (const float*)d_in[4];
  const float* Wv = (const float*)d_in[5];
  const float* bv = (const float*)d_in[6];
  const float* Wp = (const float*)d_in[7];
  const float* bp = (const float*)d_in[8];
  float* out = (float*)d_out;

  // ws layout (halfs): xh(=ab after QKV) | q | k | vT | Wq,Wk,Wv,Wp (f16)
  half_t* ws = (half_t*)d_ws;
  const size_t SEG = (size_t)TOKENS * DIM;     // 16.7M halfs
  const size_t WSEG = (size_t)DIM * DIM;       // 262144 halfs
  half_t* xh  = ws;            // reused as attention output after QKV gemms
  half_t* qb  = ws + SEG;
  half_t* kb  = ws + 2 * SEG;
  half_t* vb  = ws + 3 * SEG;
  half_t* wqh = ws + 4 * SEG;
  half_t* wkh = wqh + WSEG;
  half_t* wvh = wqh + 2 * WSEG;
  half_t* wph = wqh + 3 * WSEG;

  dim3 block(256);
  // one-time f16 conversions
  cvt_f32_f16<<<dim3(TOKENS * DIM / 2048), block, 0, stream>>>(x, xh, TOKENS * DIM);
  cvt_f32_f16<<<dim3(DIM * DIM / 2048), block, 0, stream>>>(Wq, wqh, DIM * DIM);
  cvt_f32_f16<<<dim3(DIM * DIM / 2048), block, 0, stream>>>(Wk, wkh, DIM * DIM);
  cvt_f32_f16<<<dim3(DIM * DIM / 2048), block, 0, stream>>>(Wv, wvh, DIM * DIM);
  cvt_f32_f16<<<dim3(DIM * DIM / 2048), block, 0, stream>>>(Wp, wph, DIM * DIM);

  dim3 grid(TOKENS / 128, DIM / 64);
  gemm512<<<grid, block, 0, stream>>>(xh, wqh, bq, qb, nullptr, 0);
  gemm512<<<grid, block, 0, stream>>>(xh, wkh, bk, kb, nullptr, 0);
  gemm512<<<grid, block, 0, stream>>>(xh, wvh, bv, vb, nullptr, 1);

  attn_win<<<dim3(TOKENS / WINSZ * NHEADS), block, 0, stream>>>(qb, kb, vb, xh);

  gemm512<<<grid, block, 0, stream>>>(xh, wph, bp, nullptr, out, 2);
}
